// SpatialAttentionModule_309237646179
// MI455X (gfx1250) — compile-verified, matrix operand order checked by reading, not run on hardware
//
#include <hip/hip_runtime.h>
#include <hip/hip_bf16.h>

// ---------------------------------------------------------------------------
// CDNA5 / gfx1250 wave32 WMMA implementation of the low-rank spatial attention
// module.  All heavy math runs through v_wmma_f32_16x16x32_f16 (f32 accum).
// Intermediates are stored f16 in d_ws to halve HBM traffic (23.3 TB/s chip,
// the problem is traffic-dominated: ~9 GFLOP vs ~128 MB of streamed data).
// K2 additionally uses the gfx1250 async global->LDS path (ASYNCcnt) when the
// toolchain exposes the builtins.
// ---------------------------------------------------------------------------

#define BB 16
#define NN 4096
#define HH 256
#define HEADS 4
#define HD 64
#define RK 8
#define EPSF 1e-8f

#if defined(__has_builtin)
#if __has_builtin(__builtin_amdgcn_global_load_async_to_lds_b128) && \
    __has_builtin(__builtin_amdgcn_s_wait_asynccnt)
#define HAVE_ASYNC_LDS 1
#endif
#endif

#ifndef HAVE_ASYNC_LDS
#define HAVE_ASYNC_LDS 0
#endif

typedef __attribute__((ext_vector_type(16))) _Float16 v16h;
typedef __attribute__((ext_vector_type(8)))  _Float16 v8h;
typedef __attribute__((ext_vector_type(8)))  float    v8f;
typedef __attribute__((ext_vector_type(4)))  int      v4i;

#if HAVE_ASYNC_LDS
// Address-space-qualified pointer types matching the async builtin signature:
// (v4i addrspace(1)*, v4i addrspace(3)*, imm offset, imm cpol)
typedef __attribute__((address_space(1))) v4i* as1_v4i_p;
typedef __attribute__((address_space(3))) v4i* as3_v4i_p;
#endif

union AF  { v16h v; v8h h2[2]; _Float16 h[16]; };
union HF8 { v8h  v; _Float16 h[8]; };

__device__ __forceinline__ float elu1(float t) {
  return t > 0.0f ? t + 1.0f : __expf(t);
}

__device__ __forceinline__ v8f wmma16(v16h a, v16h b, v8f c) {
  // D = A(16x32) * B(32x16) + C, f32 accumulate
  return __builtin_amdgcn_wmma_f32_16x16x32_f16(
      /*neg_a=*/false, a, /*neg_b=*/false, b,
      /*c_mod=*/(short)0, c, /*reuse_a=*/false, /*reuse_b=*/false);
}

// A fragment (row-major f16, ld in halves) or B fragment (col-major f16 view).
// Lane<16: rc = rc0+lane, K slots {0..7, 16..23}; lane>=16: K {8..15, 24..31}.
__device__ __forceinline__ v16h frag_f16(const _Float16* base, int ld, int rc0, int kk) {
  int lane = threadIdx.x & 31;
  int rc = rc0 + (lane & 15);
  int k0 = kk + ((lane < 16) ? 0 : 8);
  const _Float16* p = base + (size_t)rc * ld + k0;
  AF f;
  f.h2[0] = *(const v8h*)(p);
  f.h2[1] = *(const v8h*)(p + 16);
  return f.v;
}

// Same but source is f32 (converted), optional elu1.
template <bool ELU>
__device__ __forceinline__ v16h frag_f32(const float* base, int ld, int rc0, int kk) {
  int lane = threadIdx.x & 31;
  int rc = rc0 + (lane & 15);
  int k0 = kk + ((lane < 16) ? 0 : 8);
  const float* p = base + (size_t)rc * ld + k0;
  AF f;
#pragma unroll
  for (int i = 0; i < 8; ++i) {
    float a = p[i], b = p[16 + i];
    if (ELU) { a = elu1(a); b = elu1(b); }
    f.h[i] = (_Float16)a;
    f.h[8 + i] = (_Float16)b;
  }
  return f.v;
}

// A fragment with only K=0..7 valid (rank-8), rest zero. Row-major f32 src.
template <bool ELU>
__device__ __forceinline__ v16h frag_r8_rows(const float* base, int ld, int row0) {
  int lane = threadIdx.x & 31;
  AF f;
#pragma unroll
  for (int i = 0; i < 16; ++i) f.h[i] = (_Float16)0.0f;
  if (lane < 16) {
    const float* p = base + (size_t)(row0 + lane) * ld;
#pragma unroll
    for (int i = 0; i < 8; ++i) {
      float a = p[i];
      if (ELU) a = elu1(a);
      f.h[i] = (_Float16)a;
    }
  }
  return f.v;
}

// B fragment (col-major view) with only K=0..7 valid. base[(col)*ld + k] f32.
template <bool ELU>
__device__ __forceinline__ v16h frag_r8_cols(const float* base, int ld, int col0) {
  int lane = threadIdx.x & 31;
  AF f;
#pragma unroll
  for (int i = 0; i < 16; ++i) f.h[i] = (_Float16)0.0f;
  if (lane < 16) {
    const float* p = base + (size_t)(col0 + lane) * ld;
#pragma unroll
    for (int i = 0; i < 8; ++i) {
      float a = p[i];
      if (ELU) a = elu1(a);
      f.h[i] = (_Float16)a;
    }
  }
  return f.v;
}

// B fragment from v[h] laid out [r][4096]: element (k=r, col=m), K=0..7 valid.
__device__ __forceinline__ v16h frag_v_col(const float* vh, int col0) {
  int lane = threadIdx.x & 31;
  AF f;
#pragma unroll
  for (int i = 0; i < 16; ++i) f.h[i] = (_Float16)0.0f;
  if (lane < 16) {
    const float* p = vh + (col0 + lane);
#pragma unroll
    for (int i = 0; i < 8; ++i) f.h[i] = (_Float16)p[(size_t)i * NN];
  }
  return f.v;
}

// A fragment from vp = elu1(v[h]) [8][4096]: rows 0..7 valid, rows 8..15 zero.
__device__ __forceinline__ v16h frag_vp(const float* vh, int n0) {
  int lane = threadIdx.x & 31;
  int row = lane & 15;
  AF f;
#pragma unroll
  for (int i = 0; i < 16; ++i) f.h[i] = (_Float16)0.0f;
  if (row < 8) {
    int k0 = n0 + ((lane < 16) ? 0 : 8);
    const float* p = vh + (size_t)row * NN + k0;
#pragma unroll
    for (int i = 0; i < 8; ++i) {
      f.h[i]     = (_Float16)elu1(p[i]);
      f.h[8 + i] = (_Float16)elu1(p[16 + i]);
    }
  }
  return f.v;
}

// ---------------------------------------------------------------------------
// K0: vpsum[h*8+r] = sum_n elu1(v[h,r,n])
// ---------------------------------------------------------------------------
__global__ void k0_prep(const float* __restrict__ v, float* __restrict__ vpsum) {
  __shared__ float part[256];
  int tid = threadIdx.x;
  int row = tid >> 3;          // 0..31  (h*8+r)
  int p   = tid & 7;
  const float* src = v + (size_t)row * NN + p * 512;
  float s = 0.0f;
  for (int i = 0; i < 512; ++i) s += elu1(src[i]);
  part[tid] = s;
  __syncthreads();
  if (p == 0) {
    float t = 0.0f;
#pragma unroll
    for (int q = 0; q < 8; ++q) t += part[(row << 3) + q];
    vpsum[row] = t;
  }
}

// ---------------------------------------------------------------------------
// K1: qkv = (x @ W1 + b1) @ W2 + b2 ; split/elu ; write qp [bh][n][d],
//     kpT/valT [bh][d][n] (transposed f16 for contiguous WMMA frags later).
// grid: 512 blocks x 256 thr, wave handles 16 tokens.
// ---------------------------------------------------------------------------
__global__ void k1_qkv(const float* __restrict__ x,
                       const float* __restrict__ W1, const float* __restrict__ b1,
                       const float* __restrict__ W2, const float* __restrict__ b2,
                       _Float16* __restrict__ qp, _Float16* __restrict__ kpT,
                       _Float16* __restrict__ valT) {
  __shared__ __align__(16) _Float16 w1t[32 * 256];   // [j][k] W1^T, j padded to 32
  __shared__ __align__(16) _Float16 w2t[768 * 32];   // [c][k] W2^T, k padded to 32
  __shared__ __align__(16) _Float16 y1s[8][16 * 32]; // per-wave y1 staging
  int tid = threadIdx.x;

  for (int idx = tid; idx < 256 * 24; idx += 256) {
    int k = idx / 24, j = idx % 24;
    w1t[j * 256 + k] = (_Float16)W1[k * 24 + j];
  }
  for (int idx = tid; idx < 256 * 8; idx += 256) {
    int k = idx / 8, j = 24 + (idx % 8);
    w1t[j * 256 + k] = (_Float16)0.0f;
  }
  for (int idx = tid; idx < 24 * 768; idx += 256) {
    int k = idx / 768, c = idx % 768;
    w2t[c * 32 + k] = (_Float16)W2[idx];
  }
  for (int idx = tid; idx < 768 * 8; idx += 256) {
    int c = idx / 8, k = 24 + (idx % 8);
    w2t[c * 32 + k] = (_Float16)0.0f;
  }
  __syncthreads();

  int wv = tid >> 5, lane = tid & 31;
  int tok0 = (blockIdx.x * 8 + wv) * 16;     // global token base (b*N+n)
  int off8 = (lane < 16) ? 0 : 8;

  // GEMM1: [16 x 256] x [256 x 32]
  v8f acc0 = {}, acc1 = {};
  for (int kk = 0; kk < 256; kk += 32) {
    v16h a = frag_f32<false>(x, 256, tok0, kk);
    acc0 = wmma16(a, frag_f16(w1t, 256, 0, kk), acc0);
    acc1 = wmma16(a, frag_f16(w1t, 256, 16, kk), acc1);
  }
  // stage y1 = acc + b1 (cols>=24 forced 0) into LDS as f16
  {
    _Float16* y = &y1s[wv][0];
    int j0 = lane & 15;
#pragma unroll
    for (int i = 0; i < 8; ++i) {
      int m = off8 + i;
      y[m * 32 + j0] = (_Float16)(acc0[i] + b1[j0]);
      int j1 = 16 + j0;
      float v1 = (j1 < 24) ? (acc1[i] + b1[j1]) : 0.0f;
      y[m * 32 + j1] = (_Float16)v1;
    }
  }
  __syncthreads();

  int b  = tok0 >> 12;       // token / 4096
  int n0 = tok0 & 4095;
  v16h a2 = frag_f16(&y1s[wv][0], 32, 0, 0);

  for (int c0 = 0; c0 < 768; c0 += 16) {
    v8f acc = {};
    acc = wmma16(a2, frag_f16(w2t, 32, c0, 0), acc);
    int c = c0 + (lane & 15);
    float bias = b2[c];
    if (c0 < 256) {            // q -> elu1 -> qp [bh][n][d] (row major)
      int head = c >> 6, d = c & 63;
      _Float16* dst = qp + ((size_t)(b * HEADS + head) * NN + n0) * HD + d;
#pragma unroll
      for (int i = 0; i < 8; ++i) {
        int m = off8 + i;
        dst[(size_t)m * HD] = (_Float16)elu1(acc[i] + bias);
      }
    } else if (c0 < 512) {     // k -> elu1 -> kpT [bh][d][n]
      int cc = c - 256;
      int head = cc >> 6, d = cc & 63;
      HF8 o;
#pragma unroll
      for (int i = 0; i < 8; ++i) o.h[i] = (_Float16)elu1(acc[i] + bias);
      *(v8h*)(kpT + ((size_t)(b * HEADS + head) * HD + d) * NN + n0 + off8) = o.v;
    } else {                   // val -> valT [bh][d][n]
      int cc = c - 512;
      int head = cc >> 6, d = cc & 63;
      HF8 o;
#pragma unroll
      for (int i = 0; i < 8; ++i) o.h[i] = (_Float16)(acc[i] + bias);
      *(v8h*)(valT + ((size_t)(b * HEADS + head) * HD + d) * NN + n0 + off8) = o.v;
    }
  }
}

// ---------------------------------------------------------------------------
// K1b: z[bh*N+n] = 1 / (sum_d kp[bh,d,n] + eps)
// ---------------------------------------------------------------------------
__global__ void k1b_z(const _Float16* __restrict__ kpT, float* __restrict__ z) {
  int idx = blockIdx.x * blockDim.x + threadIdx.x;   // 0 .. B*HEADS*N-1
  int bh = idx >> 12;
  int n  = idx & 4095;
  const _Float16* p = kpT + (size_t)bh * HD * NN + n;
  float s = 0.0f;
#pragma unroll 8
  for (int d = 0; d < HD; ++d) s += (float)p[(size_t)d * NN];
  z[idx] = 1.0f / (s + EPSF);
}

// ---------------------------------------------------------------------------
// K2: per (b,h): kvT[e][d] = sum_n kp[n,d]*val[n,e]  (16 waves x 16x16 tiles)
//     tmpT[e][r] (padded to 32) = sum_n elu1(v[r,n])*val[n,e]  (waves 0..3)
// grid: 64 blocks x 512 thr
// Main loop uses gfx1250 async global->LDS staging (3 slots, ASYNCcnt) when
// available; falls back to direct global WMMA-fragment loads otherwise.
// ---------------------------------------------------------------------------
__global__ void k2_kv(const _Float16* __restrict__ kpT, const _Float16* __restrict__ valT,
                      const float* __restrict__ v, float* __restrict__ kvT,
                      float* __restrict__ tmpT) {
  int bh = blockIdx.x;
  int h  = bh & 3;
  int tid = threadIdx.x, wv = tid >> 5, lane = tid & 31;
  const _Float16* kpb = kpT + (size_t)bh * HD * NN;
  const _Float16* vab = valT + (size_t)bh * HD * NN;
  int di = (wv >> 2) * 16;   // rows of kv (d)
  int ej = (wv & 3) * 16;    // cols of kv (e)
  int off8 = (lane < 16) ? 0 : 8;

  v8f acc = {};

#if HAVE_ASYNC_LDS
  // 3-slot async pipeline: per iteration stage kp[64][32] and val[64][32]
  // (4KB each) via one b128 async transfer per thread.
  __shared__ __align__(16) _Float16 lds_kp[3 * 64 * 32];
  __shared__ __align__(16) _Float16 lds_val[3 * 64 * 32];
  {
    bool isval = tid >= 256;
    int row = (tid & 255) >> 2;            // 0..63
    int q   = tid & 3;                     // 16B quarter of a 32-half row
    const _Float16* gsrc = (isval ? vab : kpb) + (size_t)row * NN + q * 8;
    _Float16* lbase = (isval ? lds_val : lds_kp) + row * 32 + q * 8;

    // pre-issue iteration 0
    __builtin_amdgcn_global_load_async_to_lds_b128(
        (as1_v4i_p)(gsrc), (as3_v4i_p)(lbase), 0, 0);

    for (int it = 0; it < 128; ++it) {
      if (it + 1 < 128) {
        int slot = (it + 1) % 3;
        __builtin_amdgcn_global_load_async_to_lds_b128(
            (as1_v4i_p)(gsrc + (size_t)(it + 1) * 32),
            (as3_v4i_p)(lbase + slot * 2048), 0, 0);
        __builtin_amdgcn_s_wait_asynccnt(1);   // slot `it` landed, next in flight
      } else {
        __builtin_amdgcn_s_wait_asynccnt(0);
      }
      __syncthreads();                          // cross-wave visibility
      int slot = it % 3;
      v16h a = frag_f16(lds_kp + slot * 2048, 32, di, 0);
      v16h b = frag_f16(lds_val + slot * 2048, 32, ej, 0);
      acc = wmma16(a, b, acc);
    }
  }
#else
  for (int n = 0; n < NN; n += 32) {
    __builtin_prefetch(kpb + (size_t)di * NN + n + 512, 0, 1);
    __builtin_prefetch(vab + (size_t)ej * NN + n + 512, 0, 1);
    v16h a = frag_f16(kpb, NN, di, n);     // A: kp^T row=d, K=n
    v16h b = frag_f16(vab, NN, ej, n);     // B: val col-major (valT), col=e
    acc = wmma16(a, b, acc);
  }
#endif

  // store transposed: kvT[e][d]
  {
    int e = ej + (lane & 15);
    *(v8f*)(kvT + (size_t)bh * HD * HD + (size_t)e * HD + di + off8) = acc;
  }

  if (wv < 4) {              // tmp: [8 x 4096] @ [4096 x 64]
    int et = wv * 16;
    const float* vh = v + (size_t)h * RK * NN;
    v8f t = {};
    for (int n = 0; n < NN; n += 32) {
      __builtin_prefetch(vab + (size_t)et * NN + n + 512, 0, 1);
      v16h a = frag_vp(vh, n);
      v16h b = frag_f16(vab, NN, et, n);
      t = wmma16(a, b, t);
    }
    float* tb = tmpT + (size_t)bh * HD * 32;
    int e = et + (lane & 15);
    *(v8f*)(tb + (size_t)e * 32 + off8) = t;          // r = 0..15 (8..15 are 0)
    v8f zz = {};
    *(v8f*)(tb + (size_t)e * 32 + 16 + off8) = zz;    // pad r = 16..31
  }
}

// ---------------------------------------------------------------------------
// K3: per (b,h,n-tile): attn^T = kv * qp^T ; bias^T = tmp * up^T
//     O[b,n, h*64+e] = z_n*attn + bias/(denom_n+eps)   (f16)
// grid: 2048 blocks x 256 thr (wave = one 16-token tile)
// ---------------------------------------------------------------------------
__global__ void k3_attn(const _Float16* __restrict__ qp, const float* __restrict__ kvT,
                        const float* __restrict__ tmpT, const float* __restrict__ u,
                        const float* __restrict__ vpsum, const float* __restrict__ z,
                        _Float16* __restrict__ O) {
  int tid = threadIdx.x, wv = tid >> 5, lane = tid & 31;
  int gid = blockIdx.x * 8 + wv;        // 0..16383
  int bh = gid >> 8;
  int ntile = gid & 255;
  int b = bh >> 2, h = bh & 3;
  int n0 = ntile * 16;
  int off8 = (lane < 16) ? 0 : 8;

  const _Float16* qpb = qp + (size_t)bh * NN * HD;
  const float* kvb = kvT + (size_t)bh * HD * HD;
  const float* tb  = tmpT + (size_t)bh * HD * 32;
  const float* ub  = u + (size_t)h * NN * RK;

  int n = n0 + (lane & 15);             // this lane's output column (token)
  float zn = z[(size_t)bh * NN + n];
  float den = EPSF;
  {
    const float* up = ub + (size_t)n * RK;
#pragma unroll
    for (int r = 0; r < RK; ++r) den += elu1(up[r]) * vpsum[h * RK + r];
  }
  float invden = 1.0f / den;

  v16h bq0 = frag_f16(qpb, HD, n0, 0);    // B = qp (col-major view), K=d 0..31
  v16h bq1 = frag_f16(qpb, HD, n0, 32);   // K=d 32..63
  v16h bu  = frag_r8_cols<true>(ub, RK, n0);

  for (int e0 = 0; e0 < HD; e0 += 16) {
    v8f aacc = {};
    aacc = wmma16(frag_f32<false>(kvb, HD, e0, 0),  bq0, aacc);
    aacc = wmma16(frag_f32<false>(kvb, HD, e0, 32), bq1, aacc);
    v8f bacc = {};
    bacc = wmma16(frag_f32<false>(tb, 32, e0, 0), bu, bacc);
    HF8 o;
#pragma unroll
    for (int i = 0; i < 8; ++i)
      o.h[i] = (_Float16)(aacc[i] * zn + bacc[i] * invden);
    // rows of D = e, col = n (per lane): contiguous e run of 8
    *(v8h*)(O + ((size_t)b * NN + n) * HH + h * HD + e0 + off8) = o.v;
  }
}

// ---------------------------------------------------------------------------
// K4: out = (O @ Wl + bl) @ Wh + bh   (f32 output to d_out)
// grid: 512 blocks x 256 thr, wave = 16 tokens
// ---------------------------------------------------------------------------
__global__ void k4_out(const _Float16* __restrict__ O,
                       const float* __restrict__ Wl, const float* __restrict__ bl,
                       const float* __restrict__ Wh, const float* __restrict__ bhi,
                       float* __restrict__ out) {
  __shared__ __align__(16) _Float16 wlt[16 * 256];   // [j][k] Wl^T, j pad 16
  __shared__ __align__(16) _Float16 wht[256 * 32];   // [c][r] Wh^T, r pad 32
  __shared__ __align__(16) _Float16 y1s[8][16 * 32];
  int tid = threadIdx.x;

  for (int idx = tid; idx < 256 * 8; idx += 256) {
    int k = idx / 8, j = idx % 8;
    wlt[j * 256 + k] = (_Float16)Wl[idx];
  }
  for (int idx = tid; idx < 256 * 8; idx += 256) {
    int k = idx / 8, j = 8 + (idx % 8);
    wlt[j * 256 + k] = (_Float16)0.0f;
  }
  for (int idx = tid; idx < 8 * 256; idx += 256) {
    int r = idx / 256, c = idx % 256;
    wht[c * 32 + r] = (_Float16)Wh[idx];
  }
  for (int idx = tid; idx < 256 * 24; idx += 256) {
    int c = idx / 24, r = 8 + (idx % 24);
    wht[c * 32 + r] = (_Float16)0.0f;
  }
  __syncthreads();

  int wv = tid >> 5, lane = tid & 31;
  int tok0 = (blockIdx.x * 8 + wv) * 16;
  int off8 = (lane < 16) ? 0 : 8;

  v8f acc = {};
  for (int kk = 0; kk < 256; kk += 32) {
    v16h a = frag_f16(O, 256, tok0, kk);
    acc = wmma16(a, frag_f16(wlt, 256, 0, kk), acc);
  }
  // stage y1 (+bl, cols>=8 zeroed, pad K 16..31 zero)
  {
    _Float16* y = &y1s[wv][0];
    int j = lane & 15;
#pragma unroll
    for (int i = 0; i < 8; ++i) {
      int m = off8 + i;
      float val = (j < 8) ? (acc[i] + bl[j]) : 0.0f;
      y[m * 32 + j] = (_Float16)val;
    }
    HF8 zz;
#pragma unroll
    for (int i = 0; i < 8; ++i) zz.h[i] = (_Float16)0.0f;
    *(v8h*)(y + (lane & 15) * 32 + 16 + off8) = zz.v;
  }
  __syncthreads();

  v16h a2 = frag_f16(&y1s[wv][0], 32, 0, 0);
  for (int c0 = 0; c0 < 256; c0 += 16) {
    v8f acc2 = {};
    acc2 = wmma16(a2, frag_f16(wht, 32, c0, 0), acc2);
    int c = c0 + (lane & 15);
    float bias = bhi[c];
#pragma unroll
    for (int i = 0; i < 8; ++i) {
      int m = off8 + i;
      out[(size_t)(tok0 + m) * HH + c] = acc2[i] + bias;
    }
  }
}

// ---------------------------------------------------------------------------
// K5: per-block sum of |u@v| over eight 16x16 tiles (one WMMA per tile)
// grid: 32768 blocks x 256 thr -> bsum[block]
// ---------------------------------------------------------------------------
__global__ void k5_reg(const float* __restrict__ u, const float* __restrict__ v,
                       float* __restrict__ bsum) {
  __shared__ float red[256];
  int tid = threadIdx.x, wv = tid >> 5;
  long t = (long)blockIdx.x * 8 + wv;   // tile id 0 .. 4*256*256-1
  int h   = (int)(t >> 16);
  int rem = (int)(t & 65535);
  int mt = rem >> 8, nt = rem & 255;
  const float* uh = u + (size_t)h * NN * RK;
  const float* vh = v + (size_t)h * RK * NN;

  v16h a = frag_r8_rows<false>(uh, RK, mt * 16);
  v16h b = frag_v_col(vh, nt * 16);
  v8f d = {};
  d = wmma16(a, b, d);
  float s = 0.0f;
#pragma unroll
  for (int i = 0; i < 8; ++i) s += fabsf(d[i]);
  red[tid] = s;
  __syncthreads();
  for (int st = 128; st > 0; st >>= 1) {
    if (tid < st) red[tid] += red[tid + st];
    __syncthreads();
  }
  if (tid == 0) bsum[blockIdx.x] = red[0];
}

// ---------------------------------------------------------------------------
// K6: deterministic serial finalize of reg loss
// ---------------------------------------------------------------------------
__global__ void k6_fin(const float* __restrict__ bsum, const float* __restrict__ log_reg,
                       float* __restrict__ out_reg) {
  float s = 0.0f;
  for (int i = 0; i < 32768; ++i) s += bsum[i];
  out_reg[0] = __expf(log_reg[0]) * s / 67108864.0f;  // mean over 4*4096*4096
}

// ---------------------------------------------------------------------------
extern "C" void kernel_launch(void* const* d_in, const int* in_sizes, int n_in,
                              void* d_out, int out_size, void* d_ws, size_t ws_size,
                              hipStream_t stream) {
  (void)in_sizes; (void)n_in; (void)out_size; (void)ws_size;
  const float* x   = (const float*)d_in[0];
  const float* W1  = (const float*)d_in[1];
  const float* b1  = (const float*)d_in[2];
  const float* W2  = (const float*)d_in[3];
  const float* b2  = (const float*)d_in[4];
  const float* Wl  = (const float*)d_in[5];
  const float* bl  = (const float*)d_in[6];
  const float* Wh  = (const float*)d_in[7];
  const float* bhi = (const float*)d_in[8];
  const float* u   = (const float*)d_in[9];
  const float* v   = (const float*)d_in[10];
  const float* lr  = (const float*)d_in[11];
  float* out = (float*)d_out;

  char* ws = (char*)d_ws;
  size_t off = 0;
  auto take = [&](size_t bytes) -> void* {
    void* p = ws + off;
    off += (bytes + 255) & ~(size_t)255;
    return p;
  };
  const size_t NE = (size_t)BB * HEADS * NN * HD;     // 16.7M elems
  _Float16* qp   = (_Float16*)take(NE * 2);
  _Float16* kpT  = (_Float16*)take(NE * 2);
  _Float16* valT = (_Float16*)take(NE * 2);
  _Float16* Obuf = (_Float16*)take(NE * 2);
  float* zbuf  = (float*)take((size_t)BB * HEADS * NN * 4);
  float* kvT   = (float*)take((size_t)BB * HEADS * HD * HD * 4);
  float* tmpT  = (float*)take((size_t)BB * HEADS * HD * 32 * 4);
  float* vpsum = (float*)take(HEADS * RK * 4);
  float* bsum  = (float*)take(32768 * 4);

  k0_prep<<<1, 256, 0, stream>>>(v, vpsum);
  k1_qkv <<<512, 256, 0, stream>>>(x, W1, b1, W2, b2, qp, kpT, valT);
  k1b_z  <<<1024, 256, 0, stream>>>(kpT, zbuf);
  k2_kv  <<<64, 512, 0, stream>>>(kpT, valT, v, kvT, tmpT);
  k3_attn<<<2048, 256, 0, stream>>>(qp, kvT, tmpT, u, vpsum, zbuf, Obuf);
  k4_out <<<512, 256, 0, stream>>>(Obuf, Wl, bl, Wh, bhi, out);
  k5_reg <<<32768, 256, 0, stream>>>(u, v, bsum);
  k6_fin <<<1, 1, 0, stream>>>(bsum, lr, out + (size_t)BB * NN * HH);
}
